// DDCIModule_42545946034211
// MI455X (gfx1250) — matrix operand order repaired and checked by reading, not compiled, not run on hardware
//
#include <hip/hip_runtime.h>
#include <hip/hip_bf16.h>
#include <math.h>

// ---------------------------------------------------------------------------
// MI455X (gfx1250): all GEMM-shaped ops (1x1 convs, 3x3 convs as 9-tap
// implicit GEMM with TAP-MAJOR K ordering, attention QK^T and V*A^T) via
// V_WMMA_F32_16X16X32_F16. wave32, 256-thread blocks = 8 waves; block tile
// 64x256, wave tile 32x64 (2x4 WMMA tiles -> 8 v_wmma per K-chunk).
// Tap-major K (k = tap*cin + c, cin % 32 == 0) makes conv gather per-chunk
// uniform; (tap, cbase, ky, kx) are tracked incrementally across the K loop
// so no scalar divisions remain in the hot loop. Staging gathers floats into
// registers (batched loads), converts, then one packed ds_store_b64/b128.
// ---------------------------------------------------------------------------

typedef _Float16 half_t;
typedef __attribute__((ext_vector_type(16))) _Float16 v16h;
typedef __attribute__((ext_vector_type(8)))  _Float16 v8h;
typedef __attribute__((ext_vector_type(4)))  _Float16 v4h;
typedef __attribute__((ext_vector_type(8)))  float    v8f;

#define BM 64
#define BN 256
#define BK 32
#define NTHREADS 256

// fixed module dims from the reference
#define BB 16
#define CC 256
#define HH 48
#define WW 48
#define HWP (HH * WW)   // 2304 pixels per image == attention sequence length

struct GemmParams {
    // A (strided): A[m][k] = Aptr[z*a_bs + m*a_sm + k*a_sk]
    // (conv 3x3 weights use tap-major decode instead: m*a_sm + c*9 + tap)
    const float* Aptr; long long a_bs; int a_sm; int a_sk;
    // B: strided or implicit-GEMM conv gather
    const float* Bptr; long long b_bs; int b_sk; int b_sn;
    int b_conv;          // 1: conv mode (batch derived from column index n)
    int cin;             // conv input channels (K = cin * T, cin % 32 == 0)
    int in_ct, in_co;    // NCHW channel-count / channel-offset of input tensor
    int T;               // taps: 1 (1x1) or 9 (3x3)
    int dil;             // dilation (pad == dil in this model)
    // problem sizes (launches guarantee N % BN == 0, K % BK == 0)
    int M, N, K;
    // C output
    float* Cptr; long long c_bs; int c_sm; int c_sn;
    int out_conv;        // 1: NCHW store (batch from n); 0: strided (batch = blockIdx.z)
    int out_ct, out_co;
    // epilogue
    const float* bias;   // per-m bias or nullptr
    int emode;           // 0 none, 1 relu, 3 +aux1, 4 aux1*sigmoid(.)+aux2, 5 .+aux1*gate[b][m]
    const float* aux1; const float* aux2; const float* gatep;
    int aux_ct;          // channel count of aux tensors ([B][aux_ct][HW])
};

__global__ __launch_bounds__(NTHREADS)
void wmma_gemm_kernel(GemmParams p) {
    __shared__ __align__(16) half_t sA[BM * BK];   // [m][k]
    __shared__ __align__(16) half_t sB[BN * BK];   // [n][k] (transposed stage)

    const int tid  = threadIdx.x;
    const int n0   = blockIdx.x * BN;
    const int m0   = blockIdx.y * BM;
    const int z    = blockIdx.z;

    const int wave  = tid >> 5;       // 0..7
    const int lane  = tid & 31;
    const int wm    = wave >> 2;      // 0..1 : 32-row strip
    const int wn    = wave & 3;       // 0..3 : 64-col strip
    const int halfl = lane >> 4;      // half-wave select (ISA f16 frag layout)
    const int l15   = lane & 15;

    const bool convA = p.b_conv && (p.T == 9);

    v8f acc[2][4] = {};

    // incremental tap state (tap-major K: k = tap*cin + c); no divisions
    int tap = 0, cbase = 0, ky = 0, kx = 0;

    for (int k0 = 0; k0 < p.K; k0 += BK) {
        int dy = 0, dx = 0;
        if (convA) { dy = (ky - 1) * p.dil; dx = (kx - 1) * p.dil; }

        __syncthreads();

        // prefetch next weight chunk (global_prefetch_b8)
        if (k0 + BK < p.K) {
            const float* pf = &p.Aptr[(long long)z * p.a_bs +
                                      (long long)(m0 + (tid & (BM - 1))) * p.a_sm +
                                      (k0 + BK)];
            __builtin_prefetch(pf, 0, 1);
        }

        // ---- stage A: BM x BK, 4 consecutive k per thread, ds_store_b64 ----
        #pragma unroll
        for (int it = 0; it < (BM * BK) / (NTHREADS * 4); ++it) {     // 2 iters
            int u  = tid + it * NTHREADS;     // 512 quad-units
            int ml = u >> 3;                  // row (32 k = 8 quads)
            int kq = (u & 7) * 4;
            int mg  = m0 + ml;
            bool mok = mg < p.M;
            long long mrow = (long long)z * p.a_bs +
                             (long long)(mok ? mg : 0) * p.a_sm;
            float f[4];
            if (convA) {
                // weight offset = m*K + c*9 + tap, constant stride 9
                long long base = mrow + (long long)(cbase + kq) * 9 + tap;
                #pragma unroll
                for (int q = 0; q < 4; ++q) f[q] = p.Aptr[base + q * 9];
            } else {
                long long base = mrow + (long long)(k0 + kq) * p.a_sk;
                #pragma unroll
                for (int q = 0; q < 4; ++q)
                    f[q] = p.Aptr[base + (long long)q * p.a_sk];
            }
            union { v4h v; half_t h[4]; } q4;
            #pragma unroll
            for (int q = 0; q < 4; ++q) q4.h[q] = (half_t)(mok ? f[q] : 0.f);
            *(v4h*)&sA[ml * BK + kq] = q4.v;
        }

        // ---- stage B: BN x BK, 8 consecutive k per thread, ds_store_b128 ----
        #pragma unroll
        for (int it = 0; it < (BN * BK) / (NTHREADS * 8); ++it) {     // 4 iters
            int u  = tid + it * NTHREADS;     // 1024 octet-units
            int nl = u & (BN - 1);
            int k8 = (u >> 8) * 8;            // 0,8,16,24
            int ng = n0 + nl;
            float f[8];
            if (p.b_conv) {
                int b   = ng / HWP;
                int rem = ng - b * HWP;
                int y = rem / WW, x = rem - y * WW;
                int yy = y + dy, xx = x + dx;
                bool ok = (yy >= 0) & (yy < HH) & (xx >= 0) & (xx < WW);
                long long base =
                    ((long long)(b * p.in_ct + p.in_co + cbase + k8)) * HWP +
                    (ok ? yy : 0) * WW + (ok ? xx : 0);
                #pragma unroll
                for (int q = 0; q < 8; ++q) f[q] = p.Bptr[base + q * HWP];
                #pragma unroll
                for (int q = 0; q < 8; ++q) f[q] = ok ? f[q] : 0.f;
            } else {
                long long base = (long long)z * p.b_bs +
                                 (long long)(k0 + k8) * p.b_sk +
                                 (long long)ng * p.b_sn;
                #pragma unroll
                for (int q = 0; q < 8; ++q)
                    f[q] = p.Bptr[base + (long long)q * p.b_sk];
            }
            union { v8h v; half_t h[8]; } o8;
            #pragma unroll
            for (int q = 0; q < 8; ++q) o8.h[q] = (half_t)f[q];
            *(v8h*)&sB[nl * BK + k8] = o8.v;
        }
        __syncthreads();

        // ---- fragments per ISA layout ----
        // A 16x32 f16: lanes 0-15 -> K 0-7 & 16-23, lanes 16-31 -> K 8-15 & 24-31
        union frag { v16h v; v8h h[2]; } af[2];
        #pragma unroll
        for (int i = 0; i < 2; ++i) {
            int r = wm * 32 + i * 16 + l15;
            af[i].h[0] = *(const v8h*)&sA[r * BK + halfl * 8];
            af[i].h[1] = *(const v8h*)&sA[r * BK + 16 + halfl * 8];
        }
        // B 32x16 f16: lane = column, lanes 0-15 -> K 0-7 & 16-23, lanes 16-31 -> K 8-15 & 24-31
        #pragma unroll
        for (int j = 0; j < 4; ++j) {
            union frag bf;
            int c = wn * 64 + j * 16 + l15;
            bf.h[0] = *(const v8h*)&sB[c * BK + halfl * 8];
            bf.h[1] = *(const v8h*)&sB[c * BK + 16 + halfl * 8];
            acc[0][j] = __builtin_amdgcn_wmma_f32_16x16x32_f16(
                false, af[0].v, false, bf.v, (short)0, acc[0][j], false, false);
            acc[1][j] = __builtin_amdgcn_wmma_f32_16x16x32_f16(
                false, af[1].v, false, bf.v, (short)0, acc[1][j], false, false);
        }

        // incremental tap bookkeeping (scalar adds/compares only)
        cbase += BK;
        if (convA && cbase == p.cin) {
            cbase = 0; ++tap; ++kx;
            if (kx == 3) { kx = 0; ++ky; }
        }
    }

    // ---- fused epilogue + store (C/D layout: m = r + 8*halfl, n = lane&15) ----
    #pragma unroll
    for (int j = 0; j < 4; ++j) {
        int ng = n0 + wn * 64 + j * 16 + l15;
        if (ng >= p.N) continue;
        int bidx, pix;
        if (p.out_conv) { bidx = ng / HWP; pix = ng - bidx * HWP; }
        else            { bidx = z;        pix = ng; }
        #pragma unroll
        for (int i = 0; i < 2; ++i) {
            int mbase = m0 + wm * 32 + i * 16 + halfl * 8;
            #pragma unroll
            for (int r = 0; r < 8; ++r) {
                int mg = mbase + r;
                if (mg >= p.M) continue;
                float v = acc[i][j][r];
                if (p.bias) v += p.bias[mg];
                long long aidx = ((long long)(bidx * p.aux_ct + mg)) * HWP + pix;
                switch (p.emode) {
                    case 1: v = v > 0.f ? v : 0.f; break;
                    case 3: v += p.aux1[aidx]; break;
                    case 4: {
                        float g = 1.f / (1.f + __expf(-v));
                        v = p.aux1[aidx] * g + p.aux2[aidx];
                    } break;
                    case 5: v += p.aux1[aidx] * p.gatep[bidx * p.M + mg]; break;
                    default: break;
                }
                long long oidx;
                if (p.out_conv)
                    oidx = ((long long)(bidx * p.out_ct + p.out_co + mg)) * HWP + pix;
                else
                    oidx = (long long)z * p.c_bs + (long long)mg * p.c_sm +
                           (long long)ng * p.c_sn;
                p.Cptr[oidx] = v;
            }
        }
    }
}

// ---------------- small supporting kernels ----------------

__global__ __launch_bounds__(256)
void softmax_rows_kernel(float* attn, int nn) {
    __shared__ float red[256];
    float* r = attn + (long long)blockIdx.x * nn;
    int tid = threadIdx.x;
    float m = -1e30f;
    for (int j = tid; j < nn; j += 256) m = fmaxf(m, r[j]);
    red[tid] = m; __syncthreads();
    for (int s = 128; s > 0; s >>= 1) {
        if (tid < s) red[tid] = fmaxf(red[tid], red[tid + s]);
        __syncthreads();
    }
    float bm = red[0]; __syncthreads();
    float sum = 0.f;
    for (int j = tid; j < nn; j += 256) {
        float e = __expf(r[j] - bm); r[j] = e; sum += e;
    }
    red[tid] = sum; __syncthreads();
    for (int s = 128; s > 0; s >>= 1) {
        if (tid < s) red[tid] += red[tid + s];
        __syncthreads();
    }
    float inv = 1.f / red[0];
    for (int j = tid; j < nn; j += 256) r[j] *= inv;
}

__global__ __launch_bounds__(256)
void concat_x2_kernel(const float* __restrict__ x2, float* __restrict__ xcat) {
    long long i = (long long)blockIdx.x * 256 + threadIdx.x;
    if (i >= (long long)BB * CC * HWP) return;
    int b = (int)(i / ((long long)CC * HWP));
    long long rem = i - (long long)b * CC * HWP;
    xcat[((long long)b * 2 * CC + CC) * HWP + rem] = x2[i];
}

__global__ __launch_bounds__(256)
void dwconv3x3_kernel(const float* __restrict__ t, const float* __restrict__ w,
                      const float* __restrict__ bias, float* __restrict__ out) {
    long long i = (long long)blockIdx.x * 256 + threadIdx.x;
    if (i >= (long long)BB * CC * HWP) return;
    int pix = (int)(i % HWP);
    long long bc = i / HWP;
    int c = (int)(bc % CC);
    int y = pix / WW, x = pix - y * WW;
    float s = bias[c];
    for (int ky = 0; ky < 3; ++ky)
        for (int kx = 0; kx < 3; ++kx) {
            int yy = y + ky - 1, xx = x + kx - 1;
            bool ok = (yy >= 0) & (yy < HH) & (xx >= 0) & (xx < WW);
            int yyc = ok ? yy : 0, xxc = ok ? xx : 0;
            float xv = t[bc * HWP + yyc * WW + xxc];
            s += w[c * 9 + ky * 3 + kx] * (ok ? xv : 0.f);
        }
    out[i] = s;
}

__global__ __launch_bounds__(256)
void gap_kernel(const float* __restrict__ xcat, float* __restrict__ gap) {
    __shared__ float red[256];
    const float* src = xcat + (long long)blockIdx.x * HWP;   // blockIdx.x over B*2C
    float s = 0.f;
    for (int j = threadIdx.x; j < HWP; j += 256) s += src[j];
    red[threadIdx.x] = s; __syncthreads();
    for (int st = 128; st > 0; st >>= 1) {
        if (threadIdx.x < st) red[threadIdx.x] += red[threadIdx.x + st];
        __syncthreads();
    }
    if (threadIdx.x == 0) gap[blockIdx.x] = red[0] / (float)HWP;
}

__global__ __launch_bounds__(256)
void cgate_kernel(const float* __restrict__ gap, const float* __restrict__ w1,
                  const float* __restrict__ b1, const float* __restrict__ w2,
                  const float* __restrict__ b2, float* __restrict__ gate) {
    __shared__ float hid[64];
    int b = blockIdx.x, t = threadIdx.x;
    if (t < 64) {
        float s = b1[t];
        const float* g = gap + b * 2 * CC;
        for (int k = 0; k < 2 * CC; ++k) s += w1[t * 2 * CC + k] * g[k];
        hid[t] = s > 0.f ? s : 0.f;
    }
    __syncthreads();
    if (t < CC) {
        float s = b2[t];
        for (int k = 0; k < 64; ++k) s += w2[t * 64 + k] * hid[k];
        gate[b * CC + t] = 1.f / (1.f + __expf(-s));
    }
}

// ---------------- host-side orchestration ----------------

static void launch_gemm(const GemmParams& p, int Z, hipStream_t s) {
    dim3 grid((p.N + BN - 1) / BN, (p.M + BM - 1) / BM, Z);
    wmma_gemm_kernel<<<grid, dim3(NTHREADS), 0, s>>>(p);
}

extern "C" void kernel_launch(void* const* d_in, const int* in_sizes, int n_in,
                              void* d_out, int out_size, void* d_ws, size_t ws_size,
                              hipStream_t stream) {
    const float* x1    = (const float*)d_in[0];
    const float* x2    = (const float*)d_in[1];
    const float* w_sr1 = (const float*)d_in[2];  const float* b_sr1 = (const float*)d_in[3];
    const float* w_sr2 = (const float*)d_in[4];  const float* b_sr2 = (const float*)d_in[5];
    const float* w_sr3 = (const float*)d_in[6];  const float* b_sr3 = (const float*)d_in[7];
    const float* w_fuse= (const float*)d_in[8];  const float* b_fuse= (const float*)d_in[9];
    const float* wq    = (const float*)d_in[10]; const float* bq    = (const float*)d_in[11];
    const float* wk    = (const float*)d_in[12]; const float* bk    = (const float*)d_in[13];
    const float* wv    = (const float*)d_in[14]; const float* bv    = (const float*)d_in[15];
    const float* w_ce  = (const float*)d_in[16]; const float* b_ce  = (const float*)d_in[17];
    const float* w_cr  = (const float*)d_in[18]; const float* b_cr  = (const float*)d_in[19];
    const float* w_st  = (const float*)d_in[20]; const float* b_st  = (const float*)d_in[21];
    const float* w_ds1 = (const float*)d_in[22]; const float* b_ds1 = (const float*)d_in[23];
    const float* w_ds2 = (const float*)d_in[24]; const float* b_ds2 = (const float*)d_in[25];
    const float* w_dc1 = (const float*)d_in[26]; const float* b_dc1 = (const float*)d_in[27];
    const float* w_dc2 = (const float*)d_in[28]; const float* b_dc2 = (const float*)d_in[29];
    float* out = (float*)d_out;

    const long long BCHW  = (long long)BB * CC * HWP;       // 9.4M f32
    const long long B2CHW = 2 * BCHW;
    const long long NNSQ  = (long long)BB * HWP * HWP;      // attention matrix

    float* ws = (float*)d_ws;
    size_t off = 0;
    auto carve = [&](long long n) { float* p = ws + off; off += (size_t)n; return p; };
    float* mssr  = carve((long long)BB * 96 * HWP);
    float* rx1   = carve(BCHW);      // later reused as "fused"
    float* qb    = carve(BCHW);
    float* kb    = carve(BCHW);      // later reused as "t"
    float* vb    = carve(BCHW);      // later reused as "spatial_out"
    float* attn  = carve(NNSQ);
    float* ceb   = carve(B2CHW);
    float* xcat  = carve(B2CHW);
    float* gapb  = carve(BB * 2 * CC);
    float* gateb = carve(BB * CC);
    (void)ws_size; (void)in_sizes; (void)n_in; (void)out_size;

    const int Nflat = BB * HWP;   // 36864 = 144 * 256

    GemmParams p{};
    p.aux_ct = CC; p.dil = 1; p.T = 1; p.a_sk = 1; p.b_conv = 1; p.out_conv = 1;
    p.cin = CC;

    // ---- CSCA: 3 dilated 3x3 reduce convs (256 -> 32), concat into mssr[96] ----
    const float* srw[3] = { w_sr1, w_sr2, w_sr3 };
    const float* srb[3] = { b_sr1, b_sr2, b_sr3 };
    for (int s = 0; s < 3; ++s) {
        GemmParams c = p;
        c.Aptr = srw[s]; c.a_sm = CC * 9;
        c.Bptr = x1; c.cin = CC; c.in_ct = CC; c.in_co = 0; c.T = 9; c.dil = s + 1;
        c.M = 32; c.N = Nflat; c.K = CC * 9;
        c.Cptr = mssr; c.out_ct = 96; c.out_co = 32 * s;
        c.bias = srb[s]; c.emode = 0;
        launch_gemm(c, 1, stream);
    }
    // ---- fuse 1x1: 96 -> 256 ----
    { GemmParams c = p;
      c.Aptr = w_fuse; c.a_sm = 96;
      c.Bptr = mssr; c.cin = 96; c.in_ct = 96;
      c.M = CC; c.N = Nflat; c.K = 96;
      c.Cptr = rx1; c.out_ct = CC; c.bias = b_fuse; c.emode = 0;
      launch_gemm(c, 1, stream); }
    // ---- q/k/v 1x1 convs ----
    { GemmParams c = p;
      c.a_sm = CC; c.cin = CC; c.in_ct = CC; c.M = CC; c.N = Nflat; c.K = CC;
      c.out_ct = CC; c.emode = 0;
      c.Aptr = wq; c.Bptr = rx1; c.Cptr = qb; c.bias = bq; launch_gemm(c, 1, stream);
      c.Aptr = wk; c.Bptr = x2;  c.Cptr = kb; c.bias = bk; launch_gemm(c, 1, stream);
      c.Aptr = wv; c.Bptr = x2;  c.Cptr = vb; c.bias = bv; launch_gemm(c, 1, stream); }

    // ---- S = q^T k  (per batch: 2304 x 2304 x 256) ----
    { GemmParams c{};
      c.aux_ct = CC; c.cin = CC; c.T = 1;
      c.Aptr = qb; c.a_bs = (long long)CC * HWP; c.a_sm = 1;   c.a_sk = HWP;
      c.Bptr = kb; c.b_bs = (long long)CC * HWP; c.b_sk = HWP; c.b_sn = 1;
      c.b_conv = 0; c.out_conv = 0;
      c.M = HWP; c.N = HWP; c.K = CC;
      c.Cptr = attn; c.c_bs = (long long)HWP * HWP; c.c_sm = HWP; c.c_sn = 1;
      c.bias = nullptr; c.emode = 0;
      launch_gemm(c, BB, stream); }

    softmax_rows_kernel<<<dim3(BB * HWP), dim3(256), 0, stream>>>(attn, HWP);

    // ---- attended = v @ attn^T ; fused = attended + x1 (epilogue) ----
    float* fused = rx1;  // rx1 dead after q
    { GemmParams c{};
      c.aux_ct = CC; c.cin = CC; c.T = 1;
      c.Aptr = vb;   c.a_bs = (long long)CC * HWP;  c.a_sm = HWP; c.a_sk = 1;
      c.Bptr = attn; c.b_bs = (long long)HWP * HWP; c.b_sk = 1;   c.b_sn = HWP;
      c.b_conv = 0; c.out_conv = 0;
      c.M = CC; c.N = HWP; c.K = HWP;
      c.Cptr = fused; c.c_bs = (long long)CC * HWP; c.c_sm = HWP; c.c_sn = 1;
      c.bias = nullptr; c.emode = 3; c.aux1 = x1;
      launch_gemm(c, BB, stream); }

    // ---- channel gate: ce = relu(conv1x1 256->512), cr -> sigmoid gate,
    //      csca_out = fused*gate + x1 written into xcat channels [0,256) ----
    { GemmParams c = p;
      c.Aptr = w_ce; c.a_sm = CC;
      c.Bptr = fused; c.cin = CC; c.in_ct = CC;
      c.M = 2 * CC; c.N = Nflat; c.K = CC;
      c.Cptr = ceb; c.out_ct = 2 * CC; c.bias = b_ce; c.emode = 1;
      launch_gemm(c, 1, stream); }
    { GemmParams c = p;
      c.Aptr = w_cr; c.a_sm = 2 * CC;
      c.Bptr = ceb; c.cin = 2 * CC; c.in_ct = 2 * CC;
      c.M = CC; c.N = Nflat; c.K = 2 * CC;
      c.Cptr = xcat; c.out_ct = 2 * CC; c.out_co = 0;
      c.bias = b_cr; c.emode = 4; c.aux1 = fused; c.aux2 = x1;
      launch_gemm(c, 1, stream); }

    // xcat channels [256,512) = x2
    concat_x2_kernel<<<dim3((unsigned)((BCHW + 255) / 256)), dim3(256), 0, stream>>>(x2, xcat);

    // ---- CMDF spatial branch: t = relu(conv1x1 512->256), depthwise 3x3 ----
    float* tb   = kb;  // k dead after S
    float* spat = vb;  // v dead after attended
    { GemmParams c = p;
      c.Aptr = w_ds1; c.a_sm = 2 * CC;
      c.Bptr = xcat; c.cin = 2 * CC; c.in_ct = 2 * CC;
      c.M = CC; c.N = Nflat; c.K = 2 * CC;
      c.Cptr = tb; c.out_ct = CC; c.bias = b_ds1; c.emode = 1;
      launch_gemm(c, 1, stream); }
    dwconv3x3_kernel<<<dim3((unsigned)((BCHW + 255) / 256)), dim3(256), 0, stream>>>(
        tb, w_ds2, b_ds2, spat);

    // ---- CMDF channel branch: GAP(xcat) -> tiny MLP -> sigmoid gate ----
    gap_kernel<<<dim3(BB * 2 * CC), dim3(256), 0, stream>>>(xcat, gapb);
    cgate_kernel<<<dim3(BB), dim3(256), 0, stream>>>(gapb, w_dc1, b_dc1, w_dc2, b_dc2, gateb);

    // ---- static 3x3 conv on csca_out (= xcat ch [0,256)),
    //      out = static + spatial*channel_gate (epilogue 5) ----
    { GemmParams c = p;
      c.Aptr = w_st; c.a_sm = CC * 9;
      c.Bptr = xcat; c.cin = CC; c.in_ct = 2 * CC; c.in_co = 0; c.T = 9; c.dil = 1;
      c.M = CC; c.N = Nflat; c.K = CC * 9;
      c.Cptr = out; c.out_ct = CC; c.out_co = 0;
      c.bias = b_st; c.emode = 5; c.aux1 = spat; c.gatep = gateb;
      launch_gemm(c, 1, stream); }
}
